// SequenceAttention_32736240730446
// MI455X (gfx1250) — compile-verified, matrix operand order checked by reading, not run on hardware
//
#include <hip/hip_runtime.h>
#include <hip/hip_bf16.h>
#include <math.h>
#include <stdint.h>

// ---------------------------------------------------------------------------
// Problem constants: N=2048, B=4, S=1024, SFZ=384, IFZ=256, H=8, A=64, H*A=512
// ---------------------------------------------------------------------------
#define kN   2048
#define kB   4
#define kS   1024
#define kSFZ 384
#define kIFZ 256
#define kH   8
#define kA   64
#define kHA  512

typedef __bf16 bf16_t;
typedef __attribute__((ext_vector_type(16))) __bf16 v16bf;
typedef __attribute__((ext_vector_type(8)))  __bf16 v8bf;
typedef __attribute__((ext_vector_type(8)))  float  v8f;

// K-pair base index for VGPR v / lane-half (CDNA5 ISA, 16-bit 16x32):
// half 0: v0..3 -> K 0..7, v4..7 -> K 16..23 ; half 1: +8
__device__ __forceinline__ int frag_kbase(int v, int half) {
  return (v < 4 ? 2 * v : 16 + 2 * (v - 4)) + 8 * half;
}

// Stride-1 bf16 fragment: exactly two aligned 16-byte loads.
// f[0..7] = K (8*half .. 8*half+7), f[8..15] = K (16+8*half .. 16+8*half+7)
__device__ __forceinline__ v16bf ldfrag_bf16_c(const bf16_t* p, int half) {
  const v8bf lo8 = *(const v8bf*)(p + 8 * half);
  const v8bf hi8 = *(const v8bf*)(p + 16 + 8 * half);
  v16bf f;
#pragma unroll
  for (int j = 0; j < 8; ++j) { f[j] = lo8[j]; f[8 + j] = hi8[j]; }
  return f;
}

// Stride-1 f32 fragment with on-the-fly bf16 convert (for raw fp32 inputs).
__device__ __forceinline__ v16bf ldfrag_f32(const float* p, int half) {
  v16bf f;
#pragma unroll
  for (int v = 0; v < 8; ++v) {
    int kb = frag_kbase(v, half);
    f[2 * v]     = (__bf16)p[kb];
    f[2 * v + 1] = (__bf16)p[kb + 1];
  }
  return f;
}

__device__ __forceinline__ v8f wmma_bf16(v16bf a, v16bf b, v8f c) {
  // (neg_a, A, neg_b, B, c_mod, C, reuse_a, reuse_b)
  return __builtin_amdgcn_wmma_f32_16x16x32_bf16(false, a, false, b, (short)0, c,
                                                 false, false);
}

// ---------------------------------------------------------------------------
// Weight convert + transpose: Wt[n*K + k] = bf16(W[k*Nn + n])
// Makes every GEMM B-fragment a pair of contiguous 16B bf16 loads.
// ---------------------------------------------------------------------------
__global__ void convt_kernel(const float* __restrict__ W, bf16_t* __restrict__ Wt,
                             int K, int Nn) {
  int i = blockIdx.x * blockDim.x + threadIdx.x;
  if (i >= K * Nn) return;
  int k = i / Nn, n = i - k * Nn;
  Wt[(size_t)n * K + k] = (bf16_t)W[i];
}

// ---------------------------------------------------------------------------
// Wave-per-16x16-tile bf16 WMMA GEMM. Bt is [Nn,K] bf16 (col-major of B).
// MODE 1: sigmoid f32 out; 2: bf16 row-major out; 3: f32 + bias + sqrt2*resid;
// MODE 4: bf16 transposed out (Cbf[c*M + r]).
// ---------------------------------------------------------------------------
template <int MODE, bool ABF>
__global__ __launch_bounds__(256) void gemm_kernel(
    const float* Af, const bf16_t* Abf, const bf16_t* Bt,
    const float* bias, const float* resid,
    float* Cf, bf16_t* Cbf, int M, int K, int Nn)
{
  const int lane   = threadIdx.x & 31;
  const int wave   = blockIdx.x * (blockDim.x >> 5) + (threadIdx.x >> 5);
  const int ntiles = Nn >> 4;
  const int mt     = wave / ntiles;
  const int nt     = wave - mt * ntiles;
  if (mt * 16 >= M) return;
  const int half = lane >> 4;
  const int lo   = lane & 15;
  const int row  = mt * 16 + lo;   // A-fragment row
  const int col  = nt * 16 + lo;   // B-fragment column

  v8f acc = {};
  for (int k0 = 0; k0 < K; k0 += 32) {
    v16bf a;
    if constexpr (ABF) {
      a = ldfrag_bf16_c(Abf + (size_t)row * K + k0, half);
    } else {
      if (k0 + 32 < K) __builtin_prefetch(Af + (size_t)row * K + k0 + 32, 0, 1);
      a = ldfrag_f32(Af + (size_t)row * K + k0, half);
    }
    v16bf b = ldfrag_bf16_c(Bt + (size_t)col * K + k0, half);
    acc = wmma_bf16(a, b, acc);
  }

#pragma unroll
  for (int i = 0; i < 8; ++i) {
    const int r = mt * 16 + i + 8 * half;   // C layout: VGPR i -> row i+8*half
    const int c = nt * 16 + lo;
    float v = acc[i];
    if (bias) v += bias[c];
    if constexpr (MODE == 1) v = 1.0f / (1.0f + __expf(-v));
    if constexpr (MODE == 3) v += 1.41421356237309515f * resid[(size_t)r * Nn + c];
    if constexpr (MODE == 2)      Cbf[(size_t)r * Nn + c] = (bf16_t)v;
    else if constexpr (MODE == 4) Cbf[(size_t)c * M + r]  = (bf16_t)v;
    else                          Cf [(size_t)r * Nn + c] = v;
  }
}

// ---------------------------------------------------------------------------
// Fused attention: one block = 16 nodes x 1 head, 4 waves.
// q tile staged in LDS via global_load_async_to_lds_b128 (+s_wait_asynccnt),
// score tiles per candidate batch in [bmin,bmax] (sorted batch => usually 1),
// softmax in LDS, feat from transposed V (stride-1 fragments), gate, bf16 out.
// ---------------------------------------------------------------------------
__global__ __launch_bounds__(128) void attn_kernel(
    const bf16_t* qbf, const bf16_t* kbf, const bf16_t* vt,
    const float* gate, const int* batch, const int* mask,
    float* s01, bf16_t* gfeat)
{
  __shared__ bf16_t sc[16][kS];    // 32 KB: scores -> exp -> weights
  __shared__ bf16_t qs[16][kA];    // 2 KB: staged q tile
  __shared__ float  red[16][8];
  __shared__ float  rowstat[16];
  __shared__ int    rowb[16];

  const int h    = blockIdx.x & (kH - 1);
  const int n0   = (blockIdx.x >> 3) * 16;
  const int tid  = threadIdx.x;
  const int wave = tid >> 5;
  const int lane = tid & 31;
  const int half = lane >> 4;
  const int lo   = lane & 15;

  if (tid < 16) rowb[tid] = batch[n0 + tid];

  // ---- async stage of the 16x64 q tile into LDS (CDNA5 async-to-LDS) ----
  {
    const int r = tid >> 3, ch = tid & 7;            // 128 threads x 16B = 2 KB
    const bf16_t* src = qbf + (size_t)(n0 + r) * kHA + h * kA + ch * 8;
    unsigned long long ga = (unsigned long long)(uintptr_t)src;
    unsigned int la = (unsigned int)(uintptr_t)&qs[r][ch * 8];
    asm volatile("global_load_async_to_lds_b128 %0, %1, off"
                 :: "v"(la), "v"(ga) : "memory");
  }
  asm volatile("s_wait_asynccnt 0x0" ::: "memory");
  __syncthreads();

  const int bmin = rowb[0];
  const int bmax = rowb[15];

  // q A-fragments are loop-invariant: hoist them out of the s/b loops.
  const v16bf qa0 = ldfrag_bf16_c(&qs[lo][0],  half);
  const v16bf qa1 = ldfrag_bf16_c(&qs[lo][32], half);

  // ---- phase 1: scores = (Q K^T)/sqrt(A), per-row batch select ----
  for (int st = wave; st < kS / 16; st += 4) {
    const int s0 = st * 16;
    float sel[8];
#pragma unroll
    for (int i = 0; i < 8; ++i) sel[i] = 0.0f;

    for (int b = bmin; b <= bmax; ++b) {
      const bf16_t* krow = kbf + ((size_t)b * kS + s0 + lo) * kHA + h * kA;
      v8f c = {};
      c = wmma_bf16(qa0, ldfrag_bf16_c(krow, half),      c);
      c = wmma_bf16(qa1, ldfrag_bf16_c(krow + 32, half), c);
#pragma unroll
      for (int i = 0; i < 8; ++i)
        if (rowb[i + 8 * half] == b) sel[i] = c[i] * 0.125f;  // 1/sqrt(64)
    }

    const int scol = s0 + lo;
#pragma unroll
    for (int i = 0; i < 8; ++i) {
      const int r  = i + 8 * half;
      const int rb = rowb[r];
      const float raw = sel[i];
      if (h < 2) s01[((size_t)(n0 + r) * kS + scol) * 2 + h] = raw;  // pre-mask
      sc[r][scol] = (bf16_t)(mask[rb * kS + scol] ? raw : -3.0e38f);
    }
  }
  __syncthreads();

  // ---- phase 2: masked row softmax over S ----
  {
    const int r = tid >> 3, j = tid & 7;   // 16 rows x 8 threads
    float mx = -3.0e38f;
    for (int s = j; s < kS; s += 8) mx = fmaxf(mx, (float)sc[r][s]);
    red[r][j] = mx;
    __syncthreads();
    if (j == 0) {
      float m2 = red[r][0];
#pragma unroll
      for (int t = 1; t < 8; ++t) m2 = fmaxf(m2, red[r][t]);
      rowstat[r] = m2;
    }
    __syncthreads();
    const float rm = rowstat[r];
    float sum = 0.0f;
    for (int s = j; s < kS; s += 8) {
      float sv = (float)sc[r][s];
      float e  = (sv > -1.0e37f) ? __expf(sv - rm) : 0.0f;   // e * mask
      sc[r][s] = (bf16_t)e;
      sum += e;
    }
    red[r][j] = sum;
    __syncthreads();
    if (j == 0) {
      float s2 = 0.0f;
#pragma unroll
      for (int t = 0; t < 8; ++t) s2 += red[r][t];
      rowstat[r] = s2;
    }
    __syncthreads();
    const float inv = 1.0f / (rowstat[r] + 1e-9f);
    for (int s = j; s < kS; s += 8) sc[r][s] = (bf16_t)((float)sc[r][s] * inv);
  }
  __syncthreads();

  // ---- phase 3: feat = W @ V (A from LDS, V transposed -> stride-1 frags) ----
  {
    const int ct = wave;                   // 16-col tile of A=64
    float fsel[8];
#pragma unroll
    for (int i = 0; i < 8; ++i) fsel[i] = 0.0f;

    for (int b = bmin; b <= bmax; ++b) {
      const bf16_t* vcol = vt + (size_t)(h * kA + ct * 16 + lo) * (kB * kS)
                              + (size_t)b * kS;
      v8f f = {};
      for (int kk = 0; kk < kS; kk += 32) {
        v16bf wa = ldfrag_bf16_c(&sc[lo][kk], half);     // LDS (ds_load_b128)
        v16bf vb = ldfrag_bf16_c(vcol + kk, half);       // global b128
        f = wmma_bf16(wa, vb, f);
      }
#pragma unroll
      for (int i = 0; i < 8; ++i)
        if (rowb[i + 8 * half] == b) fsel[i] = f[i];
    }
#pragma unroll
    for (int i = 0; i < 8; ++i) {
      const int n = n0 + i + 8 * half;
      const int c = h * kA + ct * 16 + lo;
      gfeat[(size_t)n * kHA + c] = (bf16_t)(gate[(size_t)n * kHA + c] * fsel[i]);
    }
  }
}

// seq_aa_logits[n,s] = scores[n,s,0] + scores[n,s,1]
__global__ void logits_kernel(const float* s01, float* logits, int total) {
  int i = blockIdx.x * blockDim.x + threadIdx.x;
  if (i < total) logits[i] = s01[2 * (size_t)i] + s01[2 * (size_t)i + 1];
}

// Wave-per-row LayerNorm over IFZ=256 (8 cols/lane, shfl_xor wave32 reduction)
__global__ __launch_bounds__(64) void ln_kernel(const float* y, const float* gamma,
                                                const float* beta, float* out) {
  const int row  = blockIdx.x * (blockDim.x >> 5) + (threadIdx.x >> 5);
  const int lane = threadIdx.x & 31;
  float vals[8];
  float s = 0.0f;
#pragma unroll
  for (int j = 0; j < 8; ++j) {
    vals[j] = y[(size_t)row * kIFZ + lane + j * 32];
    s += vals[j];
  }
#pragma unroll
  for (int off = 16; off; off >>= 1) s += __shfl_xor(s, off, 32);
  const float mu = s * (1.0f / kIFZ);
  float v = 0.0f;
#pragma unroll
  for (int j = 0; j < 8; ++j) { float d = vals[j] - mu; v += d * d; }
#pragma unroll
  for (int off = 16; off; off >>= 1) v += __shfl_xor(v, off, 32);
  const float rs = rsqrtf(v * (1.0f / kIFZ) + 1e-5f);
#pragma unroll
  for (int j = 0; j < 8; ++j) {
    const int c = lane + j * 32;
    out[(size_t)row * kIFZ + c] = (vals[j] - mu) * rs * gamma[c] + beta[c];
  }
}

// ---------------------------------------------------------------------------
extern "C" void kernel_launch(void* const* d_in, const int* in_sizes, int n_in,
                              void* d_out, int out_size, void* d_ws, size_t ws_size,
                              hipStream_t stream) {
  (void)in_sizes; (void)n_in; (void)out_size; (void)ws_size;

  const float* x     = (const float*)d_in[0];
  const float* emb   = (const float*)d_in[1];
  const int*   mask  = (const int*)d_in[2];
  const int*   batch = (const int*)d_in[3];
  const float* Wq    = (const float*)d_in[4];
  const float* Wk    = (const float*)d_in[5];
  const float* Wv    = (const float*)d_in[6];
  const float* Wg    = (const float*)d_in[7];
  const float* bg    = (const float*)d_in[8];
  const float* Wback = (const float*)d_in[9];
  const float* bback = (const float*)d_in[10];
  const float* gamma = (const float*)d_in[11];
  const float* beta  = (const float*)d_in[12];

  // workspace layout (~36 MB total; fits in the 192 MB L2)
  char* ws = (char*)d_ws;
  bf16_t* qbf   = (bf16_t*)ws; ws += (size_t)kN * kHA * sizeof(bf16_t);
  bf16_t* kbf   = (bf16_t*)ws; ws += (size_t)kB * kS * kHA * sizeof(bf16_t);
  bf16_t* vt    = (bf16_t*)ws; ws += (size_t)kB * kS * kHA * sizeof(bf16_t); // [HA, B*S]
  bf16_t* gfeat = (bf16_t*)ws; ws += (size_t)kN * kHA * sizeof(bf16_t);
  bf16_t* wqT   = (bf16_t*)ws; ws += (size_t)kHA * kIFZ * sizeof(bf16_t);
  bf16_t* wgT   = (bf16_t*)ws; ws += (size_t)kHA * kIFZ * sizeof(bf16_t);
  bf16_t* wkT   = (bf16_t*)ws; ws += (size_t)kHA * kSFZ * sizeof(bf16_t);
  bf16_t* wvT   = (bf16_t*)ws; ws += (size_t)kHA * kSFZ * sizeof(bf16_t);
  bf16_t* wbT   = (bf16_t*)ws; ws += (size_t)kIFZ * kHA * sizeof(bf16_t);
  float*  gatep = (float*)ws;  ws += (size_t)kN * kHA * sizeof(float);
  float*  s01   = (float*)ws;  ws += (size_t)kN * kS * 2 * sizeof(float);
  float*  ybuf  = (float*)ws;  ws += (size_t)kN * kIFZ * sizeof(float);

  float* out_ln     = (float*)d_out;
  float* out_logits = out_ln + (size_t)kN * kIFZ;

  auto blocks_for = [](int tiles) { return (tiles + 7) / 8; };  // 8 waves/block
  auto cblocks    = [](int n)     { return (n + 255) / 256; };

  // Pre-transpose + convert weights to bf16 column-major.
  convt_kernel<<<cblocks(kIFZ * kHA), 256, 0, stream>>>(Wq,    wqT, kIFZ, kHA);
  convt_kernel<<<cblocks(kIFZ * kHA), 256, 0, stream>>>(Wg,    wgT, kIFZ, kHA);
  convt_kernel<<<cblocks(kSFZ * kHA), 256, 0, stream>>>(Wk,    wkT, kSFZ, kHA);
  convt_kernel<<<cblocks(kSFZ * kHA), 256, 0, stream>>>(Wv,    wvT, kSFZ, kHA);
  convt_kernel<<<cblocks(kHA * kIFZ), 256, 0, stream>>>(Wback, wbT, kHA, kIFZ);

  // q = x @ Wq -> bf16 row-major
  gemm_kernel<2, false><<<blocks_for((kN / 16) * (kHA / 16)), 256, 0, stream>>>(
      x, nullptr, wqT, nullptr, nullptr, nullptr, qbf, kN, kIFZ, kHA);
  // gate = sigmoid(x @ Wg + bg) -> f32
  gemm_kernel<1, false><<<blocks_for((kN / 16) * (kHA / 16)), 256, 0, stream>>>(
      x, nullptr, wgT, bg, nullptr, gatep, nullptr, kN, kIFZ, kHA);
  // k = emb @ Wk -> bf16 row-major
  gemm_kernel<2, false><<<blocks_for((kB * kS / 16) * (kHA / 16)), 256, 0, stream>>>(
      emb, nullptr, wkT, nullptr, nullptr, nullptr, kbf, kB * kS, kSFZ, kHA);
  // v = emb @ Wv -> bf16 TRANSPOSED ([HA, B*S]) for stride-1 phase-3 fragments
  gemm_kernel<4, false><<<blocks_for((kB * kS / 16) * (kHA / 16)), 256, 0, stream>>>(
      emb, nullptr, wvT, nullptr, nullptr, nullptr, vt, kB * kS, kSFZ, kHA);
  // fused attention
  attn_kernel<<<(kN / 16) * kH, 128, 0, stream>>>(qbf, kbf, vt, gatep, batch,
                                                  mask, s01, gfeat);
  // seq_aa_logits
  const int total = kN * kS;
  logits_kernel<<<(total + 255) / 256, 256, 0, stream>>>(s01, out_logits, total);
  // y = sqrt2 * x + gfeat @ Wback + bback
  gemm_kernel<3, true><<<blocks_for((kN / 16) * (kIFZ / 16)), 256, 0, stream>>>(
      nullptr, gfeat, wbT, bback, x, ybuf, nullptr, kN, kHA, kIFZ);
  // LayerNorm -> out
  ln_kernel<<<kN / 2, 64, 0, stream>>>(ybuf, gamma, beta, out_ln);
}
